// HGNN_Q_Info_13666585936297
// MI455X (gfx1250) — compile-verified, matrix operand order checked by reading, not run on hardware
//
#include <hip/hip_runtime.h>
#include <hip/hip_bf16.h>

typedef __bf16 bf16_t;
typedef __attribute__((ext_vector_type(16))) __bf16 v16bf;
typedef __attribute__((ext_vector_type(8)))  float  v8f;
typedef __attribute__((ext_vector_type(4)))  unsigned int v4u;

#define LH 20          // window length
#define DD 36          // input dim
#define HH 64          // hidden dim
#define G3 192         // 3*H gate width
#define TILE_T 16      // timesteps per wave (WMMA M)
#define WAVES 4
#define BLOCK_T (TILE_T * WAVES)          // 64 timesteps per block
#define EROWS (BLOCK_T + LH - 1)          // 83 embedding rows per block

#define WMMA_BF16(A, B, C) \
  __builtin_amdgcn_wmma_f32_16x16x32_bf16(false, (A), false, (B), (short)0, (C), false, false)

// Guard-free fragment load from a 16B-aligned bf16 row (K stride 64, chunk base
// pre-added). Per ISA 7.12.2 the 8 fragment dwords are two contiguous 16-byte
// runs selected by the lane half: dwords [half*4 .. half*4+3] and [8+half*4 ..].
__device__ __forceinline__ v16bf frag_ld(const bf16_t* rowk0, int half) {
  const unsigned int* p = (const unsigned int*)rowk0;
  v4u lo = *(const v4u*)(p + half * 4);
  v4u hi = *(const v4u*)(p + 8 + half * 4);
  union { v16bf bf; unsigned int u[8]; } f;
#pragma unroll
  for (int i = 0; i < 4; ++i) { f.u[i] = lo[i]; f.u[4 + i] = hi[i]; }
  return f.bf;
}

__device__ __forceinline__ float fsig(float x) {
  return __builtin_amdgcn_rcpf(1.0f + __expf(-x));     // v_exp_f32 + v_rcp_f32
}
__device__ __forceinline__ float ftanh(float x) {
  return 1.0f - 2.0f * __builtin_amdgcn_rcpf(1.0f + __expf(2.0f * x));
}

__global__ __launch_bounds__(128, 1) void gru_window_wmma_kernel(
    const float* __restrict__ emb,  const float* __restrict__ Wih,
    const float* __restrict__ Whh,  const float* __restrict__ bih,
    const float* __restrict__ bhh,  const float* __restrict__ W1,
    const float* __restrict__ b1v,  const float* __restrict__ W2,
    const float* __restrict__ b2v,  const int* __restrict__ dones,
    float* __restrict__ out, int n) {
  // All operand tiles zero-padded to K stride 64 so fragment loads need no guards.
  __shared__ __align__(16) bf16_t EmbT[EROWS * 64];     // sliding embedding rows
  __shared__ __align__(16) bf16_t WihT[G3 * 64];        // W_ih, cols 36..63 = 0
  __shared__ __align__(16) bf16_t WhhT[G3 * 64];        // W_hh
  __shared__ __align__(16) bf16_t W1T[HH * 64];         // W1
  __shared__ __align__(16) bf16_t W2T[16 * 64];         // W2, rows 4..15 = 0
  __shared__ __align__(16) bf16_t Hb[WAVES][TILE_T * 64]; // hidden state (bf16)
  __shared__ unsigned vmask[BLOCK_T];

  const int tid  = threadIdx.x;
  const int w    = tid >> 5;
  const int lane = tid & 31;
  const int half = lane >> 4;
  const int l16  = lane & 15;
  const int t_base = blockIdx.x * BLOCK_T;

  // ---- cooperative staging ----
  for (int i = tid; i < EROWS * 64; i += 128) {
    int r = i >> 6, c = i & 63;
    long gr = (long)t_base - (LH - 1) + r;
    float v = (c < DD && gr >= 0 && gr < n) ? emb[gr * DD + c] : 0.0f;
    EmbT[i] = (bf16_t)v;
  }
  for (int i = tid; i < G3 * 64; i += 128) {
    int j = i >> 6, c = i & 63;
    WihT[i] = (bf16_t)((c < DD) ? Wih[j * DD + c] : 0.0f);
    WhhT[i] = (bf16_t)Whh[i];
  }
  for (int i = tid; i < HH * 64; i += 128) W1T[i] = (bf16_t)W1[i];
  for (int i = tid; i < 16 * 64; i += 128) {
    int j = i >> 6, c = i & 63;
    W2T[i] = (bf16_t)((j < 4) ? W2[j * 64 + c] : 0.0f);
  }
  {
    bf16_t* hz = &Hb[0][0];
    for (int i = tid; i < WAVES * TILE_T * 64; i += 128) hz[i] = (bf16_t)0.0f;
  }
  if (tid < BLOCK_T) {
    int t = t_base + tid;
    unsigned vm = 0u;
    if (t < n) {
      int ep = 0;  // episode start if a done occurred inside the lookback
      for (int j = t - (LH - 1); j <= t - 1; ++j)
        if (j >= 0 && dones[j] != 0) ep = j + 1;
      for (int l = 0; l < LH; ++l) {
        int g = t - (LH - 1) + l;
        if (g >= 0 && g >= ep) vm |= (1u << l);
      }
    }
    vmask[tid] = vm;
  }
  __syncthreads();

  // per-lane bias columns (column index = l16 within each 16-wide tile)
  float bihv[12], bhhv[12];
#pragma unroll
  for (int i = 0; i < 12; ++i) {
    bihv[i] = bih[i * 16 + l16];
    bhhv[i] = bhh[i * 16 + l16];
  }
  unsigned vmr[8];
#pragma unroll
  for (int r = 0; r < 8; ++r) vmr[r] = vmask[w * TILE_T + r + 8 * half];

  bf16_t* Hw = Hb[w];
  const bf16_t* hrow = Hw + l16 * 64;                 // A-operand row for h
  const bf16_t* erow0 = EmbT + (w * TILE_T + l16) * 64;

  // ---- 20-step batched GRU (kept rolled; 48 WMMA per iteration) ----
#pragma unroll 1
  for (int l = 0; l < LH; ++l) {
    const bf16_t* erow = erow0 + l * 64;
    v16bf ae0 = frag_ld(erow, half);
    v16bf ae1 = frag_ld(erow + 32, half);
    v16bf ah0 = frag_ld(hrow, half);
    v16bf ah1 = frag_ld(hrow + 32, half);

#pragma unroll
    for (int g = 0; g < 4; ++g) {
      const bf16_t* wir = WihT + (g * 16 + l16) * 64;
      const bf16_t* whr = WhhT + (g * 16 + l16) * 64;
      v8f accr = {}, accz = {}, accni = {}, accnh = {};
      // r gate (cols g*16): x- and h- contributions merged
      accr = WMMA_BF16(ae0, frag_ld(wir, half), accr);
      accr = WMMA_BF16(ae1, frag_ld(wir + 32, half), accr);
      accr = WMMA_BF16(ah0, frag_ld(whr, half), accr);
      accr = WMMA_BF16(ah1, frag_ld(whr + 32, half), accr);
      // z gate (cols 64 + g*16)
      accz = WMMA_BF16(ae0, frag_ld(wir + 64 * 64, half), accz);
      accz = WMMA_BF16(ae1, frag_ld(wir + 64 * 64 + 32, half), accz);
      accz = WMMA_BF16(ah0, frag_ld(whr + 64 * 64, half), accz);
      accz = WMMA_BF16(ah1, frag_ld(whr + 64 * 64 + 32, half), accz);
      // n gate (cols 128 + g*16): x/h parts separate (r scales only h-part)
      accni = WMMA_BF16(ae0, frag_ld(wir + 128 * 64, half), accni);
      accni = WMMA_BF16(ae1, frag_ld(wir + 128 * 64 + 32, half), accni);
      accnh = WMMA_BF16(ah0, frag_ld(whr + 128 * 64, half), accnh);
      accnh = WMMA_BF16(ah1, frag_ld(whr + 128 * 64 + 32, half), accnh);

      // elementwise GRU update for this 16-column slab
      bf16_t* hcol = Hw + (8 * half) * 64 + g * 16 + l16;
#pragma unroll
      for (int r = 0; r < 8; ++r) {
        float rr = fsig(accr[r] + bihv[g] + bhhv[g]);
        float zz = fsig(accz[r] + bihv[4 + g] + bhhv[4 + g]);
        float nn = ftanh(accni[r] + bihv[8 + g] + rr * (accnh[r] + bhhv[8 + g]));
        float hp = (float)hcol[r * 64];
        float hnew = (1.0f - zz) * nn + zz * hp;
        bool vok = (vmr[r] >> l) & 1u;
        hcol[r * 64] = (bf16_t)(vok ? hnew : hp);
      }
    }
  }

  // ---- final MLP: relu(h @ W1^T + b1) @ W2^T + b2 ----
  {
    v16bf a0 = frag_ld(hrow, half);
    v16bf a1 = frag_ld(hrow + 32, half);
    v8f hid[4];
#pragma unroll
    for (int jt = 0; jt < 4; ++jt) {
      const bf16_t* w1r = W1T + (jt * 16 + l16) * 64;
      v8f acc = {};
      acc = WMMA_BF16(a0, frag_ld(w1r, half), acc);
      acc = WMMA_BF16(a1, frag_ld(w1r + 32, half), acc);
      hid[jt] = acc;
    }
#pragma unroll
    for (int jt = 0; jt < 4; ++jt) {
      float bb = b1v[jt * 16 + l16];
      bf16_t* hcol = Hw + (8 * half) * 64 + jt * 16 + l16;
#pragma unroll
      for (int r = 0; r < 8; ++r) {
        float v = hid[jt][r] + bb;
        hcol[r * 64] = (bf16_t)(v > 0.0f ? v : 0.0f);
      }
    }
    v16bf c0 = frag_ld(hrow, half);
    v16bf c1 = frag_ld(hrow + 32, half);
    const bf16_t* w2r = W2T + l16 * 64;
    v8f acc = {};
    acc = WMMA_BF16(c0, frag_ld(w2r, half), acc);
    acc = WMMA_BF16(c1, frag_ld(w2r + 32, half), acc);
    if (l16 < 4) {
      float bb = b2v[l16];
#pragma unroll
      for (int r = 0; r < 8; ++r) {
        int m = r + 8 * half;
        int t = t_base + w * TILE_T + m;
        if (t < n) out[t * 4 + l16] = acc[r] + bb;
      }
    }
  }
}

extern "C" void kernel_launch(void* const* d_in, const int* in_sizes, int n_in,
                              void* d_out, int out_size, void* d_ws, size_t ws_size,
                              hipStream_t stream) {
  (void)n_in; (void)d_ws; (void)ws_size; (void)out_size;
  const float* emb  = (const float*)d_in[0];
  const float* Wih  = (const float*)d_in[1];
  const float* Whh  = (const float*)d_in[2];
  const float* bih  = (const float*)d_in[3];
  const float* bhh  = (const float*)d_in[4];
  const float* W1   = (const float*)d_in[5];
  const float* b1   = (const float*)d_in[6];
  const float* W2   = (const float*)d_in[7];
  const float* b2   = (const float*)d_in[8];
  const int*   dons = (const int*)d_in[9];
  float* out = (float*)d_out;
  int n = in_sizes[0] / DD;
  int grid = (n + BLOCK_T - 1) / BLOCK_T;
  gru_window_wmma_kernel<<<dim3(grid), dim3(128), 0, stream>>>(
      emb, Wih, Whh, bih, bhh, W1, b1, W2, b2, dons, out, n);
}
